// MoDLayer_81166291960282
// MI455X (gfx1250) — matrix operand order repaired and checked by reading, not compiled, not run on hardware
//
#include <hip/hip_runtime.h>

// Problem constants (from reference)
#define BB 4
#define NN 4096
#define DD 1024
#define HH 16
#define CC 1024          // capacity = N * 0.25
#define DH 64            // head dim
#define BC (BB * CC)     // 4096 rows in capacity buffer
#define LD 1024          // leading dimension of EVERY GEMM operand here

typedef __attribute__((ext_vector_type(16))) __bf16 bf16x16;
typedef __attribute__((ext_vector_type(8)))  float  f32x8;

union Frag32B {
    uint4   u[2];
    bf16x16 v;
};

// ---------------------------------------------------------------------------
// 1) Router: scores[b,n] = token[b,n,:] . w_router ; per-block z-loss partial
// ---------------------------------------------------------------------------
__global__ void router_kernel(const float* __restrict__ tok,
                              const float* __restrict__ wr,
                              float* __restrict__ scores,
                              float* __restrict__ zpart) {
    __shared__ float sh[8];
    int token = blockIdx.x * 8 + (threadIdx.x >> 5);   // 8 waves = 8 tokens
    int lane  = threadIdx.x & 31;
    const float* t = tok + (size_t)token * DD;
    float s = 0.f;
    #pragma unroll 4
    for (int i = lane; i < DD; i += 32) s += t[i] * wr[i];
    #pragma unroll
    for (int off = 16; off > 0; off >>= 1) s += __shfl_xor(s, off, 32);
    if (lane == 0) {
        scores[token] = s;
        sh[threadIdx.x >> 5] = s * s;
    }
    __syncthreads();
    if (threadIdx.x == 0) {
        float z = 0.f;
        #pragma unroll
        for (int i = 0; i < 8; ++i) z += sh[i];
        zpart[blockIdx.x] = z;
    }
}

__global__ void zreduce_kernel(const float* __restrict__ zpart,
                               float* __restrict__ zout) {
    __shared__ float sh[8];
    int lane = threadIdx.x & 31, wv = threadIdx.x >> 5;
    float s = 0.f;
    for (int i = threadIdx.x; i < 2048; i += 256) s += zpart[i];
    #pragma unroll
    for (int off = 16; off > 0; off >>= 1) s += __shfl_xor(s, off, 32);
    if (lane == 0) sh[wv] = s;
    __syncthreads();
    if (threadIdx.x == 0) {
        float z = 0.f;
        #pragma unroll
        for (int i = 0; i < 8; ++i) z += sh[i];
        zout[0] = z / (float)(BB * NN);
    }
}

// ---------------------------------------------------------------------------
// 2) Rank-based top-C selection: slot = #{j : s_j > s_i or (s_j==s_i && j<i)}
//    Selected set identical to jax.lax.top_k; slot order is a permutation of
//    top_k's, which is harmless (attention is permutation-equivariant).
// ---------------------------------------------------------------------------
__global__ void rank_kernel(const float* __restrict__ scores,
                            int* __restrict__ slot,
                            int* __restrict__ idx) {
    __shared__ float sc[NN];
    int b = blockIdx.x;
    const float* s = scores + (size_t)b * NN;
    for (int i = threadIdx.x; i < NN; i += 1024) sc[i] = s[i];
    __syncthreads();
    for (int i = threadIdx.x; i < NN; i += 1024) {
        float si = sc[i];
        int r = 0;
        for (int j = 0; j < NN; ++j) {
            float sj = sc[j];
            r += (sj > si) || ((sj == si) && (j < i));
        }
        slot[(size_t)b * NN + i] = (r < CC) ? r : -1;
        if (r < CC) idx[(size_t)b * CC + r] = i;
    }
}

// ---------------------------------------------------------------------------
// 3) Gather selected tokens -> bf16 capacity buffer xb [B*C, D].
//    Token row is staged through LDS with the CDNA5 async-to-LDS path
//    (ASYNCcnt-tracked), then read back via ds_load for the bf16 convert.
// ---------------------------------------------------------------------------
__global__ void gather_kernel(const float* __restrict__ tok,
                              const int* __restrict__ idx,
                              __bf16* __restrict__ xb) {
    __shared__ float4 sbuf[256];            // 4 KB = one token row (fp32)
    int bc = blockIdx.x;                    // 0..4095
    int b  = bc >> 10;
    int c  = bc & 1023;
    int n  = idx[(size_t)b * CC + c];
    const float4* src = (const float4*)(tok + ((size_t)b * NN + n) * DD);
    int i = threadIdx.x;                    // 256 threads * 16 B = 4 KB

    // LDS byte offset = low 32 bits of the flat shared address (LDS aperture
    // maps addr[31:0] directly to the LDS offset).
    unsigned ldsOff = (unsigned)(size_t)(&sbuf[i]);
    const float4* gsrc = src + i;
    asm volatile("global_load_async_to_lds_b128 %0, %1, off"
                 :: "v"(ldsOff), "v"(gsrc) : "memory");
    asm volatile("s_wait_asynccnt 0x0" ::: "memory");

    float4 v = sbuf[i];                     // ds_load_b128
    __bf16* dst = xb + (size_t)bc * DD;
    dst[i * 4 + 0] = (__bf16)v.x;
    dst[i * 4 + 1] = (__bf16)v.y;
    dst[i * 4 + 2] = (__bf16)v.z;
    dst[i * 4 + 3] = (__bf16)v.w;
}

// ---------------------------------------------------------------------------
// 4) Weight prep: wT[w][n*D + k] = bf16(w[k*D + n])  (transposed, bf16)
// ---------------------------------------------------------------------------
__global__ void wprep_kernel(const float* __restrict__ w0,
                             const float* __restrict__ w1,
                             const float* __restrict__ w2,
                             const float* __restrict__ w3,
                             __bf16* __restrict__ wT) {
    size_t t = (size_t)blockIdx.x * 256 + threadIdx.x; // 0 .. 4M-1
    int w = (int)(t >> 20);
    size_t e = t & 1048575u;
    int k = (int)(e >> 10);
    int n = (int)(e & 1023u);
    const float* src = (w == 0) ? w0 : (w == 1) ? w1 : (w == 2) ? w2 : w3;
    wT[(size_t)w * 1048576 + (size_t)n * DD + k] = (__bf16)src[e];
}

// ---------------------------------------------------------------------------
// 5) Register-blocked WMMA bf16 GEMM:  D(M,N) = A(M,K) * Bt(N,K)^T, fp32 acc.
//    Leading dimension is a compile-time 1024 for both operands, so the four
//    16-row subtile deltas (32768 B) become literal instruction offsets, and
//    the whole K-step needs just two lane-varying byte offsets.
//    Each wave owns a 64x64 tile = 4x4 WMMA subtiles. Per 32-wide K step:
//    load 4 B fragments once, then stream 4 A fragments, 4 WMMAs each ->
//    16 v_wmma per 16 b128 loads, peak live set ~170 VGPRs.
//    __launch_bounds__(256,1) lifts the occupancy-driven VGPR cap (no spills).
//    mode 0: fp32 row-major   mode 1: bf16 row-major
//    mode 2: bf16 transposed, batched (vT layout: [b][n][m_local])
// ---------------------------------------------------------------------------
__global__ __launch_bounds__(256, 1)
void gemm_bf16_wmma(const __bf16* __restrict__ A,
                    const __bf16* __restrict__ Bt,
                    void* __restrict__ out, int ldo,
                    int M, int N, int K, int mode,
                    int rowsPerBatch, long long outBatchStride) {
    int wavesPerBlock = blockDim.x >> 5;
    int wave = blockIdx.x * wavesPerBlock + (threadIdx.x >> 5);
    int lane = threadIdx.x & 31;
    int tilesN = N >> 6;                    // 64-wide tiles
    int totalTiles = (M >> 6) * tilesN;
    if (wave >= totalTiles) return;
    int tm = (wave / tilesN) << 6;
    int tn = (wave % tilesN) << 6;
    int r16  = lane & 15;
    int half = lane >> 4;

    // A lane row = lane%16 ; K-halves 0..7 & 16..23 (+8 for lanes 16..31)
    const __bf16* aP = A  + (size_t)(tm + r16) * LD + half * 8;
    // B lane col = lane%16 ; K-halves 0..7 & 16..23 (+8 for lanes 16..31)
    const __bf16* bP = Bt + (size_t)(tn + r16) * LD + half * 8;

    f32x8 acc[4][4] = {};
    for (int k0 = 0; k0 < K; k0 += 32) {
        __builtin_prefetch(aP + k0 + 256, 0, 1);
        __builtin_prefetch(bP + k0 + 256, 0, 1);
        Frag32B fb[4];
        #pragma unroll
        for (int j = 0; j < 4; ++j) {
            const __bf16* pb = bP + j * (16 * LD) + k0;   // +32768 B literal
            fb[j].u[0] = *(const uint4*)pb;
            fb[j].u[1] = *(const uint4*)(pb + 16);
        }
        #pragma unroll
        for (int i = 0; i < 4; ++i) {
            const __bf16* pa = aP + i * (16 * LD) + k0;   // +32768 B literal
            Frag32B fa;
            fa.u[0] = *(const uint4*)pa;
            fa.u[1] = *(const uint4*)(pa + 16);
            #pragma unroll
            for (int j = 0; j < 4; ++j)
                acc[i][j] = __builtin_amdgcn_wmma_f32_16x16x32_bf16(
                    false, fa.v, false, fb[j].v,
                    (short)0, acc[i][j], false, false);
        }
    }

    #pragma unroll
    for (int i = 0; i < 4; ++i) {
        #pragma unroll
        for (int j = 0; j < 4; ++j) {
            int col = tn + j * 16 + r16;
            if (mode == 0) {
                float* o = (float*)out;
                #pragma unroll
                for (int r = 0; r < 8; ++r) {
                    int m = tm + i * 16 + r + 8 * half;
                    o[(size_t)m * ldo + col] = acc[i][j][r];
                }
            } else if (mode == 1) {
                __bf16* o = (__bf16*)out;
                #pragma unroll
                for (int r = 0; r < 8; ++r) {
                    int m = tm + i * 16 + r + 8 * half;
                    o[(size_t)m * ldo + col] = (__bf16)acc[i][j][r];
                }
            } else {
                __bf16* o = (__bf16*)out;
                #pragma unroll
                for (int r = 0; r < 8; ++r) {
                    int m  = tm + i * 16 + r + 8 * half;
                    int b  = m / rowsPerBatch;
                    int ml = m - b * rowsPerBatch;
                    o[(size_t)b * outBatchStride + (size_t)col * ldo + ml] =
                        (__bf16)acc[i][j][r];
                }
            }
        }
    }
}

// ---------------------------------------------------------------------------
// 6) Row softmax with 1/sqrt(dh) scaling; fp32 in, bf16 out. 256 thr / row.
// ---------------------------------------------------------------------------
__global__ void softmax_kernel(const float* __restrict__ a,
                               __bf16* __restrict__ o) {
    __shared__ float red[8];
    int row = blockIdx.x;
    const float* ar = a + (size_t)row * CC;
    __bf16* orow = o + (size_t)row * CC;
    int lane = threadIdx.x & 31, wv = threadIdx.x >> 5;
    const float scale = 0.125f;     // 1/sqrt(64)
    float x0 = ar[threadIdx.x +   0] * scale;
    float x1 = ar[threadIdx.x + 256] * scale;
    float x2 = ar[threadIdx.x + 512] * scale;
    float x3 = ar[threadIdx.x + 768] * scale;
    float m = fmaxf(fmaxf(x0, x1), fmaxf(x2, x3));
    #pragma unroll
    for (int off = 16; off > 0; off >>= 1) m = fmaxf(m, __shfl_xor(m, off, 32));
    if (lane == 0) red[wv] = m;
    __syncthreads();
    float gm = red[0];
    #pragma unroll
    for (int i = 1; i < 8; ++i) gm = fmaxf(gm, red[i]);
    __syncthreads();
    float e0 = __expf(x0 - gm), e1 = __expf(x1 - gm);
    float e2 = __expf(x2 - gm), e3 = __expf(x3 - gm);
    float s = e0 + e1 + e2 + e3;
    #pragma unroll
    for (int off = 16; off > 0; off >>= 1) s += __shfl_xor(s, off, 32);
    if (lane == 0) red[wv] = s;
    __syncthreads();
    float gs = 0.f;
    #pragma unroll
    for (int i = 0; i < 8; ++i) gs += red[i];
    float inv = 1.0f / gs;
    orow[threadIdx.x +   0] = (__bf16)(e0 * inv);
    orow[threadIdx.x + 256] = (__bf16)(e1 * inv);
    orow[threadIdx.x + 512] = (__bf16)(e2 * inv);
    orow[threadIdx.x + 768] = (__bf16)(e3 * inv);
}

// ---------------------------------------------------------------------------
// 7) Combine: out[b,n,:] = selected ? sigmoid(score) * o[b,slot,:] : 0
// ---------------------------------------------------------------------------
__global__ void combine_kernel(const float* __restrict__ outo,
                               const float* __restrict__ scores,
                               const int* __restrict__ slot,
                               float* __restrict__ out) {
    int bn = blockIdx.x;            // 0..16383
    int b  = bn >> 12;              // N = 4096
    int s  = slot[bn];
    float4* dst = (float4*)(out + (size_t)bn * DD);
    int i = threadIdx.x;            // 256 threads * float4 = 1024 floats
    if (s < 0) {
        float4 z = {0.f, 0.f, 0.f, 0.f};
        dst[i] = z;
    } else {
        float g = 1.0f / (1.0f + __expf(-scores[bn]));
        const float4* src =
            (const float4*)(outo + ((size_t)(b * CC + s)) * DD);
        float4 v = src[i];
        v.x *= g; v.y *= g; v.z *= g; v.w *= g;
        dst[i] = v;
    }
}

// ---------------------------------------------------------------------------
// Host launcher
// ---------------------------------------------------------------------------
extern "C" void kernel_launch(void* const* d_in, const int* in_sizes, int n_in,
                              void* d_out, int out_size, void* d_ws, size_t ws_size,
                              hipStream_t stream) {
    const float* tok = (const float*)d_in[0];
    const float* wr  = (const float*)d_in[1];
    const float* wq  = (const float*)d_in[2];
    const float* wk  = (const float*)d_in[3];
    const float* wv  = (const float*)d_in[4];
    const float* wo  = (const float*)d_in[5];
    float* out = (float*)d_out;

    // ---- workspace layout (256-B aligned, total ~70 MB) ----
    char* wsp = (char*)d_ws;
    auto alloc = [&](size_t bytes) -> char* {
        char* p = wsp;
        wsp += (bytes + 255) & ~(size_t)255;
        return p;
    };
    float*  scores = (float*)alloc((size_t)BB * NN * 4);         // 64 KB
    float*  zpart  = (float*)alloc(2048 * 4);                    //  8 KB
    int*    slot   = (int*)  alloc((size_t)BB * NN * 4);         // 64 KB
    int*    idx    = (int*)  alloc((size_t)BB * CC * 4);         // 16 KB
    __bf16* wT     = (__bf16*)alloc((size_t)4 * DD * DD * 2);    //  8 MB
    __bf16* xb     = (__bf16*)alloc((size_t)BC * DD * 2);        //  8 MB
    __bf16* qb     = (__bf16*)alloc((size_t)BC * DD * 2);        //  8 MB
    __bf16* kb     = (__bf16*)alloc((size_t)BC * DD * 2);        //  8 MB
    __bf16* vT     = (__bf16*)alloc((size_t)BB * DD * CC * 2);   //  8 MB
    float*  attf   = (float*) alloc((size_t)CC * CC * 4);        //  4 MB
    __bf16* attb   = (__bf16*)alloc((size_t)CC * CC * 2);        //  2 MB
    __bf16* ob     = (__bf16*)alloc((size_t)BC * DD * 2);        //  8 MB
    float*  outo   = (float*) alloc((size_t)BC * DD * 4);        // 16 MB

    // 1) router + z-loss
    router_kernel<<<2048, 256, 0, stream>>>(tok, wr, scores, zpart);
    zreduce_kernel<<<1, 256, 0, stream>>>(zpart, out + (size_t)BB * NN * DD);

    // 2) top-C selection via ranking
    rank_kernel<<<BB, 1024, 0, stream>>>(scores, slot, idx);

    // 3) gather + bf16 convert (async-to-LDS staging)
    gather_kernel<<<BC, 256, 0, stream>>>(tok, idx, xb);

    // 4) weight transpose + bf16 convert (4 * 1M elements)
    wprep_kernel<<<16384, 256, 0, stream>>>(wq, wk, wv, wo, wT);

    // 5) QKV projections: [4096 x 1024] x [1024 x 1024]
    //    64x64 tiles: (4096/64)*(1024/64) = 1024 waves -> 128 blocks
    gemm_bf16_wmma<<<128, 256, 0, stream>>>(xb, wT + 0 * DD * DD,
                                            qb, DD, BC, DD, DD, 1, 0, 0);
    gemm_bf16_wmma<<<128, 256, 0, stream>>>(xb, wT + 1 * DD * DD,
                                            kb, DD, BC, DD, DD, 1, 0, 0);
    gemm_bf16_wmma<<<128, 256, 0, stream>>>(xb, wT + 2 * DD * DD,
                                            vT, CC, BC, DD, DD, 2, CC,
                                            (long long)DD * CC);

    // 6) attention per (batch, head)
    for (int b = 0; b < BB; ++b) {
        for (int h = 0; h < HH; ++h) {
            const __bf16* qh = qb + (size_t)b * CC * DD + h * DH;
            const __bf16* kh = kb + (size_t)b * CC * DD + h * DH;
            // scores = q @ k^T : M=1024, N=1024, K=64 -> 256 waves -> 32 blks
            gemm_bf16_wmma<<<32, 256, 0, stream>>>(qh, kh,
                                                   attf, CC, CC, CC, DH,
                                                   0, 0, 0);
            softmax_kernel<<<CC, 256, 0, stream>>>(attf, attb);
            // o_head = att @ v : M=1024, N=64, K=1024 -> 16 waves -> 2 blocks
            const __bf16* vh = vT + (size_t)b * DD * CC + (size_t)h * DH * CC;
            __bf16* oh = ob + (size_t)b * CC * DD + h * DH;
            gemm_bf16_wmma<<<2, 256, 0, stream>>>(attb, vh,
                                                  oh, DD, CC, DH, CC,
                                                  1, 0, 0);
        }
    }

    // 7) output projection: [4096 x 1024] x [1024 x 1024] -> fp32
    gemm_bf16_wmma<<<128, 256, 0, stream>>>(ob, wT + 3 * DD * DD,
                                            outo, DD, BC, DD, DD, 0, 0, 0);

    // 8) gated scatter back to [B, N, D] (zeros for unselected tokens)
    combine_kernel<<<BB * NN, 256, 0, stream>>>(outo, scores, slot, out);
}
